// MoELayer_87462714016471
// MI455X (gfx1250) — compile-verified, matrix operand order checked by reading, not run on hardware
//
#include <hip/hip_runtime.h>
#include <hip/hip_bf16.h>
#include <float.h>
#include <math.h>

// ---------------- problem constants ----------------
#define T_TOK 4096
#define H_DIM 1024
#define I_DIM 1024
#define E_EXP 16
#define NG 4
#define EPG 4          // E / N_GROUP
#define RSCALE 1.0f

// GEMM tiling: 256 threads = 8 wave32, 128x64 tile, K-step 32, double-buffered LDS
#define TM 128
#define TN 64
#define KT 32
#define LDT 40         // padded row stride (elements): 80B rows, 16B-aligned, low-conflict

typedef __attribute__((ext_vector_type(16))) __bf16 v16bf;
typedef __attribute__((ext_vector_type(8)))  float  v8f;

union BFrag { v16bf v; unsigned short h[16]; unsigned int d[8]; };

__device__ __forceinline__ unsigned short f2bf(float f) {
  unsigned u = __float_as_uint(f);
  unsigned r = u + 0x7FFFu + ((u >> 16) & 1u);   // round-to-nearest-even
  return (unsigned short)(r >> 16);
}
__device__ __forceinline__ unsigned pack2bf(float lo, float hi) {
  return (unsigned)f2bf(lo) | ((unsigned)f2bf(hi) << 16);
}

// K index of the low element of dword i (ISA 7.12.2, 16-bit A/B 16x32 layout)
__device__ __forceinline__ int kmap(int i, int hi) {
  return (i < 4) ? (hi * 8 + 2 * i) : (16 + hi * 8 + 2 * (i - 4));
}

__device__ __forceinline__ v16bf load_frag(const unsigned short* rowp, int hi) {
  BFrag f;
#pragma unroll
  for (int i = 0; i < 8; ++i)
    f.d[i] = *(const unsigned int*)(rowp + kmap(i, hi));
  return f.v;
}

__device__ __forceinline__ v8f wmma_bf16(v16bf a, v16bf b, v8f c) {
  return __builtin_amdgcn_wmma_f32_16x16x32_bf16(false, a, false, b, (short)0, c,
                                                 false, false);
}

// CDNA5 async global->LDS copy (VGLOBAL GLOBAL_LOAD_ASYNC_TO_LDS_B128, ASYNCcnt).
// LDS byte address = low 32 bits of the generic (flat) shared pointer (ISA 10.2).
__device__ __forceinline__ void async_copy_b128(void* lds_dst, const void* gsrc) {
  unsigned off = (unsigned)(size_t)lds_dst;
  asm volatile("global_load_async_to_lds_b128 %0, %1, off"
               : : "v"(off), "v"(gsrc) : "memory");
}
__device__ __forceinline__ void wait_async0() {
  asm volatile("s_wait_asynccnt 0x0" ::: "memory");
}

// ---------------- kernel 0: zero counters ----------------
__global__ void k_zero(int* __restrict__ cnt, int* __restrict__ cursor) {
  if (threadIdx.x < E_EXP) cnt[threadIdx.x] = 0;
  else if (threadIdx.x < 2 * E_EXP) cursor[threadIdx.x - E_EXP] = 0;
}

// ---------------- kernel 1: x fp32 -> bf16 ----------------
__global__ void k_cvt_x(const float* __restrict__ x, unsigned short* __restrict__ xb) {
  size_t i = ((size_t)blockIdx.x * blockDim.x + threadIdx.x) * 4;
  float4 v = *(const float4*)(x + i);
  unsigned long long p = (unsigned long long)pack2bf(v.x, v.y)
                       | ((unsigned long long)pack2bf(v.z, v.w) << 32);
  *(unsigned long long*)(xb + i) = p;
}

// ---------------- kernel 2: router (one wave32 per token) ----------------
__global__ __launch_bounds__(32) void k_router(
    const float* __restrict__ x, const float* __restrict__ gw, const float* __restrict__ bias,
    int* __restrict__ idx, float* __restrict__ wts, int* __restrict__ cnt) {
  __shared__ float xs[H_DIM];
  __shared__ float sSel[E_EXP], sLog[E_EXP];
  const int t = blockIdx.x, l = threadIdx.x;

#pragma unroll
  for (int i = 0; i < H_DIM / (32 * 4); ++i) {
    int o = (i * 32 + l) * 4;
    *(float4*)&xs[o] = *(const float4*)&x[(size_t)t * H_DIM + o];
  }
  __syncthreads();

  const int e = l & 15, half = l >> 4;
  const float* g  = gw + (size_t)e * H_DIM + half * (H_DIM / 2);
  const float* xr = xs + half * (H_DIM / 2);
  float acc = 0.f;
#pragma unroll 4
  for (int k = 0; k < H_DIM / 2; k += 4) {
    float4 gv = *(const float4*)(g + k);
    acc += gv.x * xr[k] + gv.y * xr[k + 1] + gv.z * xr[k + 2] + gv.w * xr[k + 3];
  }
  acc += __shfl_xor(acc, 16);
  if (half == 0) {
    float lg = 1.f / (1.f + expf(-acc));
    sLog[e] = lg;
    sSel[e] = lg + bias[e];
  }
  __syncthreads();

  if (l == 0) {
    float gs[NG];
#pragma unroll
    for (int gi = 0; gi < NG; ++gi) {
      float m1 = -FLT_MAX, m2 = -FLT_MAX;
#pragma unroll
      for (int j = 0; j < EPG; ++j) {
        float v = sSel[gi * EPG + j];
        if (v > m1) { m2 = m1; m1 = v; } else if (v > m2) m2 = v;
      }
      gs[gi] = m1 + m2;
    }
    int g0 = 0;
    for (int gi = 1; gi < NG; ++gi) if (gs[gi] > gs[g0]) g0 = gi;
    int g1 = -1;
    for (int gi = 0; gi < NG; ++gi) {
      if (gi == g0) continue;
      if (g1 < 0 || gs[gi] > gs[g1]) g1 = gi;
    }
    int i0 = 0, i1 = 0; float b0 = -FLT_MAX, b1 = -FLT_MAX;
    for (int ee = 0; ee < E_EXP; ++ee) {
      int gg = ee / EPG;
      if (gg != g0 && gg != g1) continue;
      float v = sSel[ee];
      if (v > b0)      { b1 = b0; i1 = i0; b0 = v; i0 = ee; }
      else if (v > b1) { b1 = v; i1 = ee; }
    }
    float w0 = sLog[i0], w1 = sLog[i1];
    float s = fmaxf(w0 + w1, 1e-9f);
    w0 /= s; w1 /= s;
    idx[2 * t] = i0; idx[2 * t + 1] = i1;
    wts[2 * t] = w0; wts[2 * t + 1] = w1;
    atomicAdd(&cnt[i0], 1);
    atomicAdd(&cnt[i1], 1);
  }
}

// ---------------- kernel 3: exclusive scan of counts ----------------
__global__ void k_scan(const int* __restrict__ cnt, int* __restrict__ offs) {
  if (blockIdx.x == 0 && threadIdx.x == 0) {
    int run = 0;
    for (int e = 0; e < E_EXP; ++e) { offs[e] = run; run += cnt[e]; }
    offs[E_EXP] = run;                 // == 2*T
    offs[E_EXP + 1] = run + T_TOK;     // shared expert occupies [2T, 3T)
  }
}

// ---------------- kernel 4: build sorted row lists ----------------
__global__ void k_build(const int* __restrict__ idx, const float* __restrict__ wts,
                        const int* __restrict__ offs, int* __restrict__ cursor,
                        int* __restrict__ rowlist, float* __restrict__ roww) {
  int t = blockIdx.x * blockDim.x + threadIdx.x;
  if (t >= T_TOK) return;
#pragma unroll
  for (int k = 0; k < 2; ++k) {
    int e = idx[2 * t + k];
    int pos = offs[e] + atomicAdd(&cursor[e], 1);
    rowlist[pos] = t;
    roww[pos] = wts[2 * t + k];
  }
  rowlist[2 * T_TOK + t] = t;     // shared-expert rows: identity, weight 1
  roww[2 * T_TOK + t] = 1.0f;
}

// ---------------- kernel 5: grouped GEMM1 + SwiGLU -> act (bf16) ----------------
// blockIdx = (mtile, ntile over I, expert 0..16). A tile: async global->LDS copy;
// B tile: register-staged fp32->bf16 transpose. Double buffered, 1 barrier/step.
__global__ __launch_bounds__(256) void k_gemm1(
    const unsigned short* __restrict__ xb, const float* __restrict__ Wi,
    const float* __restrict__ Ws_wi, const int* __restrict__ offs,
    const int* __restrict__ rowlist, unsigned short* __restrict__ act) {
  const int e = blockIdx.z;
  const int r0 = offs[e];
  const int count = offs[e + 1] - r0;
  const int m0 = blockIdx.x * TM;
  if (m0 >= count) return;
  int rem = count - m0; if (rem > TM) rem = TM;
  const int n0 = blockIdx.y * TN;
  const float* W = (e < E_EXP) ? (Wi + (size_t)e * H_DIM * (2 * I_DIM)) : Ws_wi;
  const int ldw = 2 * I_DIM;

  __shared__ int tks[TM];
  __shared__ __align__(16) unsigned short As[2][TM][LDT];
  __shared__ __align__(16) unsigned short Bs[2][2][TN][LDT];

  const int tid = threadIdx.x, lane = tid & 31, wv = tid >> 5;
  if (tid < TM) {
    int m = (tid < rem) ? tid : (rem - 1);
    tks[tid] = rowlist[r0 + m0 + m];
  }
  __syncthreads();

  const int arow = tid >> 1, ahf = tid & 1;           // 128 rows x 2 half-rows
  const unsigned short* asrc = xb + (size_t)tks[arow] * H_DIM + ahf * 16;
  const int bn = tid & 63, bkp = tid >> 6;            // 64 cols x 4 K-octets

  uint4 bR[2];

  auto load_a = [&](int buf, int k0) {   // async: 2 x b128 per thread = 8KB tile
    const unsigned short* s = asrc + k0;
    async_copy_b128(&As[buf][arow][ahf * 16], s);
    async_copy_b128(&As[buf][arow][ahf * 16 + 8], s + 8);
  };
  auto load_b = [&](int k0) {
#pragma unroll
    for (int h = 0; h < 2; ++h) {
      size_t col = (size_t)(h ? I_DIM : 0) + n0 + bn;
      const float* wp = W + (size_t)(k0 + bkp * 8) * ldw + col;
      unsigned* br = (unsigned*)&bR[h];
#pragma unroll
      for (int j = 0; j < 4; ++j)
        br[j] = pack2bf(wp[(size_t)(2 * j) * ldw], wp[(size_t)(2 * j + 1) * ldw]);
    }
  };
  auto store_b = [&](int buf) {
    *(uint4*)&Bs[buf][0][bn][bkp * 8] = bR[0];
    *(uint4*)&Bs[buf][1][bn][bkp * 8] = bR[1];
  };

  v8f acc[2][4];
#pragma unroll
  for (int h = 0; h < 2; ++h)
#pragma unroll
    for (int s = 0; s < 4; ++s)
#pragma unroll
      for (int j = 0; j < 8; ++j) acc[h][s][j] = 0.f;

  auto compute = [&](int buf) {
    const int mrow = wv * 16 + (lane & 15);
    const int hi = lane >> 4;
    v16bf a = load_frag(&As[buf][mrow][0], hi);
    // software-pipelined B-fragment loads: preload st+1 while WMMAing st
    v16bf bl = load_frag(&Bs[buf][0][(lane & 15)][0], hi);
    v16bf bh = load_frag(&Bs[buf][1][(lane & 15)][0], hi);
#pragma unroll
    for (int st = 0; st < 4; ++st) {
      v16bf nbl, nbh;
      if (st < 3) {
        int nrow = (st + 1) * 16 + (lane & 15);
        nbl = load_frag(&Bs[buf][0][nrow][0], hi);
        nbh = load_frag(&Bs[buf][1][nrow][0], hi);
      }
      acc[0][st] = wmma_bf16(a, bl, acc[0][st]);
      acc[1][st] = wmma_bf16(a, bh, acc[1][st]);
      bl = nbl; bh = nbh;
    }
  };

  const int nsteps = H_DIM / KT;
  load_a(0, 0);
  load_b(0);
  store_b(0);
  wait_async0();
  __syncthreads();
  for (int step = 0; step < nsteps; ++step) {
    int cur = step & 1;
    if (step + 1 < nsteps) {
      load_a(1 - cur, (step + 1) * KT);   // async copies in flight during compute
      load_b((step + 1) * KT);            // global loads in flight during compute
    }
    compute(cur);
    if (step + 1 < nsteps) store_b(1 - cur);
    wait_async0();
    __syncthreads();
  }

  // activation + store bf16. routed: silu(gate)*x_proj ; shared: silu(a)*g
  const bool shr = (e == E_EXP);
  const int nl = lane & 15, mh = (lane >> 4) * 8;
#pragma unroll
  for (int st = 0; st < 4; ++st) {
    int n = n0 + st * 16 + nl;
#pragma unroll
    for (int j = 0; j < 8; ++j) {
      int mt = wv * 16 + mh + j;
      if (mt < rem) {
        float p = acc[0][st][j], g = acc[1][st][j];
        float a_ = shr ? p : g;      // operand that gets silu
        float o_ = shr ? g : p;
        float v = (a_ / (1.f + expf(-a_))) * o_;
        act[(size_t)(r0 + m0 + mt) * I_DIM + n] = f2bf(v);
      }
    }
  }
}

// ---------------- kernel 6a: shared-expert GEMM2, writes out directly ----------------
__global__ __launch_bounds__(256) void k_gemm2_shared(
    const unsigned short* __restrict__ act, const float* __restrict__ Ws_wo,
    float* __restrict__ out) {
  const int m0 = blockIdx.x * TM;
  const int n0 = blockIdx.y * TN;
  const size_t rbase = (size_t)2 * T_TOK + m0;   // shared rows live at [2T, 3T)

  __shared__ __align__(16) unsigned short As[2][TM][LDT];
  __shared__ __align__(16) unsigned short Bs[2][TN][LDT];

  const int tid = threadIdx.x, lane = tid & 31, wv = tid >> 5;
  const int arow = tid >> 1, ahf = tid & 1;
  const unsigned short* asrc = act + (rbase + arow) * I_DIM + ahf * 16;
  const int bn = tid & 63, bkp = tid >> 6;

  uint4 bR;

  auto load_a = [&](int buf, int k0) {
    const unsigned short* s = asrc + k0;
    async_copy_b128(&As[buf][arow][ahf * 16], s);
    async_copy_b128(&As[buf][arow][ahf * 16 + 8], s + 8);
  };
  auto load_b = [&](int k0) {
    const float* wp = Ws_wo + (size_t)(k0 + bkp * 8) * H_DIM + n0 + bn;
    unsigned* br = (unsigned*)&bR;
#pragma unroll
    for (int j = 0; j < 4; ++j)
      br[j] = pack2bf(wp[(size_t)(2 * j) * H_DIM], wp[(size_t)(2 * j + 1) * H_DIM]);
  };
  auto store_b = [&](int buf) { *(uint4*)&Bs[buf][bn][bkp * 8] = bR; };

  v8f acc[4];
#pragma unroll
  for (int s = 0; s < 4; ++s)
#pragma unroll
    for (int j = 0; j < 8; ++j) acc[s][j] = 0.f;

  auto compute = [&](int buf) {
    const int mrow = wv * 16 + (lane & 15);
    const int hi = lane >> 4;
    v16bf a = load_frag(&As[buf][mrow][0], hi);
    v16bf b = load_frag(&Bs[buf][(lane & 15)][0], hi);
#pragma unroll
    for (int st = 0; st < 4; ++st) {
      v16bf nb;
      if (st < 3) nb = load_frag(&Bs[buf][(st + 1) * 16 + (lane & 15)][0], hi);
      acc[st] = wmma_bf16(a, b, acc[st]);
      b = nb;
    }
  };

  const int nsteps = I_DIM / KT;
  load_a(0, 0);
  load_b(0);
  store_b(0);
  wait_async0();
  __syncthreads();
  for (int step = 0; step < nsteps; ++step) {
    int cur = step & 1;
    if (step + 1 < nsteps) {
      load_a(1 - cur, (step + 1) * KT);
      load_b((step + 1) * KT);
    }
    compute(cur);
    if (step + 1 < nsteps) store_b(1 - cur);
    wait_async0();
    __syncthreads();
  }

  const int nl = lane & 15, mh = (lane >> 4) * 8;
#pragma unroll
  for (int st = 0; st < 4; ++st) {
    int n = n0 + st * 16 + nl;
#pragma unroll
    for (int j = 0; j < 8; ++j) {
      int t = m0 + wv * 16 + mh + j;           // shared rows are the identity map
      out[(size_t)t * H_DIM + n] = acc[st][j];
    }
  }
}

// ---------------- kernel 6b: routed GEMM2, weighted atomic scatter ----------------
__global__ __launch_bounds__(256) void k_gemm2_routed(
    const unsigned short* __restrict__ act, const float* __restrict__ Wo,
    const int* __restrict__ offs, const int* __restrict__ rowlist,
    const float* __restrict__ roww, float* __restrict__ out) {
  const int e = blockIdx.z;
  const int r0 = offs[e];
  const int count = offs[e + 1] - r0;
  const int m0 = blockIdx.x * TM;
  if (m0 >= count) return;
  int rem = count - m0; if (rem > TM) rem = TM;
  const int n0 = blockIdx.y * TN;
  const float* W = Wo + (size_t)e * I_DIM * H_DIM;

  __shared__ __align__(16) unsigned short As[2][TM][LDT];
  __shared__ __align__(16) unsigned short Bs[2][TN][LDT];

  const int tid = threadIdx.x, lane = tid & 31, wv = tid >> 5;
  const int arow = tid >> 1, ahf = tid & 1;
  const int arr = r0 + m0 + ((arow < rem) ? arow : (rem - 1));
  const unsigned short* asrc = act + (size_t)arr * I_DIM + ahf * 16;
  const int bn = tid & 63, bkp = tid >> 6;

  uint4 bR;

  auto load_a = [&](int buf, int k0) {
    const unsigned short* s = asrc + k0;
    async_copy_b128(&As[buf][arow][ahf * 16], s);
    async_copy_b128(&As[buf][arow][ahf * 16 + 8], s + 8);
  };
  auto load_b = [&](int k0) {
    const float* wp = W + (size_t)(k0 + bkp * 8) * H_DIM + n0 + bn;
    unsigned* br = (unsigned*)&bR;
#pragma unroll
    for (int j = 0; j < 4; ++j)
      br[j] = pack2bf(wp[(size_t)(2 * j) * H_DIM], wp[(size_t)(2 * j + 1) * H_DIM]);
  };
  auto store_b = [&](int buf) { *(uint4*)&Bs[buf][bn][bkp * 8] = bR; };

  v8f acc[4];
#pragma unroll
  for (int s = 0; s < 4; ++s)
#pragma unroll
    for (int j = 0; j < 8; ++j) acc[s][j] = 0.f;

  auto compute = [&](int buf) {
    const int mrow = wv * 16 + (lane & 15);
    const int hi = lane >> 4;
    v16bf a = load_frag(&As[buf][mrow][0], hi);
    v16bf b = load_frag(&Bs[buf][(lane & 15)][0], hi);
#pragma unroll
    for (int st = 0; st < 4; ++st) {
      v16bf nb;
      if (st < 3) nb = load_frag(&Bs[buf][(st + 1) * 16 + (lane & 15)][0], hi);
      acc[st] = wmma_bf16(a, b, acc[st]);
      b = nb;
    }
  };

  const int nsteps = I_DIM / KT;
  load_a(0, 0);
  load_b(0);
  store_b(0);
  wait_async0();
  __syncthreads();
  for (int step = 0; step < nsteps; ++step) {
    int cur = step & 1;
    if (step + 1 < nsteps) {
      load_a(1 - cur, (step + 1) * KT);
      load_b((step + 1) * KT);
    }
    compute(cur);
    if (step + 1 < nsteps) store_b(1 - cur);
    wait_async0();
    __syncthreads();
  }

  const int nl = lane & 15, mh = (lane >> 4) * 8;
  int tok[8]; float sc[8];
#pragma unroll
  for (int j = 0; j < 8; ++j) {
    int mt = wv * 16 + mh + j;
    if (mt < rem) {
      int r = r0 + m0 + mt;
      tok[j] = rowlist[r];
      sc[j] = RSCALE * roww[r];
    } else tok[j] = -1;
  }
#pragma unroll
  for (int st = 0; st < 4; ++st) {
    int n = n0 + st * 16 + nl;
#pragma unroll
    for (int j = 0; j < 8; ++j)
      if (tok[j] >= 0)
        atomicAdd(&out[(size_t)tok[j] * H_DIM + n], acc[st][j] * sc[j]);
  }
}

// ---------------- host launcher ----------------
extern "C" void kernel_launch(void* const* d_in, const int* in_sizes, int n_in,
                              void* d_out, int out_size, void* d_ws, size_t ws_size,
                              hipStream_t stream) {
  (void)in_sizes; (void)n_in; (void)out_size; (void)ws_size;
  const float* x      = (const float*)d_in[0];
  const float* gate_w = (const float*)d_in[1];
  const float* bias   = (const float*)d_in[2];
  const float* Wi     = (const float*)d_in[3];
  const float* Wo     = (const float*)d_in[4];
  const float* Ws_wi  = (const float*)d_in[5];
  const float* Ws_wo  = (const float*)d_in[6];
  float* out = (float*)d_out;

  char* w = (char*)d_ws;
  size_t o = 0;
  auto take = [&](size_t bytes) {
    o = (o + 255) & ~(size_t)255;
    char* p = w + o;
    o += bytes;
    return p;
  };
  unsigned short* xb  = (unsigned short*)take((size_t)T_TOK * H_DIM * 2);
  unsigned short* act = (unsigned short*)take((size_t)3 * T_TOK * I_DIM * 2);
  int*   idx     = (int*)take((size_t)T_TOK * 2 * 4);
  float* wts     = (float*)take((size_t)T_TOK * 2 * 4);
  int*   cnt     = (int*)take(E_EXP * 4);
  int*   cursor  = (int*)take(E_EXP * 4);
  int*   offs    = (int*)take((E_EXP + 2) * 4);
  int*   rowlist = (int*)take((size_t)3 * T_TOK * 4);
  float* roww    = (float*)take((size_t)3 * T_TOK * 4);

  k_zero <<<1, 32, 0, stream>>>(cnt, cursor);
  k_cvt_x<<<T_TOK * H_DIM / (256 * 4), 256, 0, stream>>>(x, xb);
  k_router<<<T_TOK, 32, 0, stream>>>(x, gate_w, bias, idx, wts, cnt);
  k_scan <<<1, 32, 0, stream>>>(cnt, offs);
  k_build<<<T_TOK / 256, 256, 0, stream>>>(idx, wts, offs, cursor, rowlist, roww);

  dim3 g1(T_TOK / TM, I_DIM / TN, E_EXP + 1);      // (32, 16, 17); inactive tiles exit
  k_gemm1<<<g1, 256, 0, stream>>>(xb, Wi, Ws_wi, offs, rowlist, act);

  dim3 g2s(T_TOK / TM, H_DIM / TN);                // shared expert covers every out elem
  k_gemm2_shared<<<g2s, 256, 0, stream>>>(act, Ws_wo, out);

  dim3 g2r(T_TOK / TM, H_DIM / TN, E_EXP);         // routed experts accumulate on top
  k_gemm2_routed<<<g2r, 256, 0, stream>>>(act, Wo, offs, rowlist, roww, out);
}
